// UV_Aggregator_6004364279884
// MI455X (gfx1250) — matrix operand order repaired and checked by reading, not compiled, not run on hardware
//
#include <hip/hip_runtime.h>

#define B_NODES 16384
#define L_HIST 50

// ws layout (ushort fragment image, then float bias block)
#define WS_W1 0
#define WS_W2 (16 * 512)
#define WS_A1 (24 * 512)
#define WS_A2 (40 * 512)
#define WS_USHORTS (48 * 512)  // 24576 ushorts = 49152 bytes
#define WS_NFLOAT 320          // b1,b2,ab1,ab2,w3 (5*64); att3_b cancels in softmax

typedef __attribute__((ext_vector_type(16))) __bf16 v16bf;
typedef __attribute__((ext_vector_type(8)))  float  v8f;

struct alignas(16) U4 { unsigned int a, b, c, d; };
union BF16Frag {
  v16bf v;
  unsigned short u[16];
  U4 q[2];
};

__device__ __forceinline__ unsigned short f2bf(float f) {
  unsigned int u = __float_as_uint(f);
  u += 0x7FFFu + ((u >> 16) & 1u);  // RNE (prep kernel only; runs once)
  return (unsigned short)(u >> 16);
}

// Packed f32x2 -> bf16x2: round-half-up + single v_perm_b32 to merge high halves.
__device__ __forceinline__ unsigned int pack_bf2(float lo, float hi) {
  unsigned int ul = __float_as_uint(lo) + 0x8000u;
  unsigned int uh = __float_as_uint(hi) + 0x8000u;
  return __builtin_amdgcn_perm(uh, ul, 0x07060302u);  // {uh[31:16], ul[31:16]}
}

__device__ __forceinline__ float relu(float x) { return x > 0.f ? x : 0.f; }

// B-fragment: pre-swizzled, 32 contiguous bytes per lane.
__device__ __forceinline__ v16bf load_bfrag(const unsigned short* fragbase, int lane) {
  BF16Frag f;
  const U4* p = reinterpret_cast<const U4*>(fragbase + lane * 16);
  f.q[0] = p[0];
  f.q[1] = p[1];
  return f.v;
}

// A-fragment from row-major bf16 activations in LDS (16-bit A 16x32 layout).
__device__ __forceinline__ v16bf load_afrag(const unsigned short* act, int stride,
                                            int kbase, int lane) {
  int m = lane & 15;
  int hf = lane >> 4;
  const unsigned short* r = act + m * stride + kbase + hf * 8;
  BF16Frag f;
  f.q[0] = *reinterpret_cast<const U4*>(r);
  f.q[1] = *reinterpret_cast<const U4*>(r + 16);
  return f.v;
}

// One 16xK @ Kx64 GEMM + bias + relu for a wave's M-tile.
// MODE 0: store bf16. MODE 1: store bf16 + f32 copy.
// MODE 2: att3 dot -> shuffle-reduced row logits (no atomics).
template <int NKT, int MODE>
__device__ __forceinline__ void wave_gemm(
    const unsigned short* actIn, int inStride,
    const unsigned short* wfrags, const float* bias,
    unsigned short* outBf, int outStride, float* outF32,
    const float* w3, float* logits, int rowBase, int lane) {
  v16bf a[NKT];
#pragma unroll
  for (int kt = 0; kt < NKT; ++kt)
    a[kt] = load_afrag(actIn, inStride, kt * 32, lane);
  const int hf = lane >> 4;
  const int cl = lane & 15;
  float dot[8];
  if constexpr (MODE == 2) {
#pragma unroll
    for (int v = 0; v < 8; ++v) dot[v] = 0.f;
  }
#pragma unroll
  for (int nt = 0; nt < 4; ++nt) {
    v8f acc = {0.f, 0.f, 0.f, 0.f, 0.f, 0.f, 0.f, 0.f};  // first WMMA gets inline C=0
#pragma unroll
    for (int kt = 0; kt < NKT; ++kt) {
      v16bf bmat = load_bfrag(wfrags + (kt * 4 + nt) * 512, lane);
      acc = __builtin_amdgcn_wmma_f32_16x16x32_bf16(
          false, a[kt], false, bmat, (short)0, acc, false, false);
    }
    const int col = nt * 16 + cl;
    const float bv = bias[col];
    if constexpr (MODE == 2) {
      const float w3v = w3[col];
#pragma unroll
      for (int v = 0; v < 8; ++v)
        dot[v] = fmaf(relu(acc[v] + bv), w3v, dot[v]);
    } else {
#pragma unroll
      for (int v = 0; v < 8; v += 2) {
        const float x0 = relu(acc[v] + bv);
        const float x1 = relu(acc[v + 1] + bv);
        const int m0 = v + hf * 8;
        const unsigned int p = pack_bf2(x0, x1);
        outBf[m0 * outStride + col] = (unsigned short)p;                // ds_store_b16
        outBf[(m0 + 1) * outStride + col] = (unsigned short)(p >> 16);  // d16_hi
        if constexpr (MODE == 1) {
          outF32[m0 * 64 + col] = x0;
          outF32[(m0 + 1) * 64 + col] = x1;
        }
      }
    }
  }
  if constexpr (MODE == 2) {
    // Sum over the 16 lanes of each half (columns); masks < 16 stay in-half.
#pragma unroll
    for (int v = 0; v < 8; ++v) {
      float d = dot[v];
      d += __shfl_xor(d, 1);
      d += __shfl_xor(d, 2);
      d += __shfl_xor(d, 4);
      d += __shfl_xor(d, 8);
      dot[v] = d;
    }
    if (cl == 0) {
#pragma unroll
      for (int v = 0; v < 8; ++v)
        logits[rowBase + v + hf * 8] = dot[v];
    }
  }
}

// ---- one-time prep: swizzle all weights to bf16 fragment image in d_ws ----
__global__ void prep_weights_kernel(
    const float* __restrict__ w1, const float* __restrict__ w2,
    const float* __restrict__ a1, const float* __restrict__ a2,
    const float* __restrict__ b1, const float* __restrict__ b2,
    const float* __restrict__ ab1, const float* __restrict__ ab2,
    const float* __restrict__ w3,
    unsigned short* __restrict__ wsu, float* __restrict__ wsf) {
  const int i = blockIdx.x * blockDim.x + threadIdx.x;
  if (i < WS_USHORTS) {
    const float* W;
    int base;
    if (i < WS_W2)      { W = w1; base = WS_W1; }
    else if (i < WS_A1) { W = w2; base = WS_W2; }
    else if (i < WS_A2) { W = a1; base = WS_A1; }
    else                { W = a2; base = WS_A2; }
    const int j = i - base;
    const int frag = j >> 9;
    const int kt = frag >> 2, nt = frag & 3;
    const int r = j & 511;
    const int ln = r >> 4, e = r & 15;
    const int k = kt * 32 + ((ln >= 16) ? 8 : 0) + (e & 7) + ((e >= 8) ? 16 : 0);
    const int n = nt * 16 + (ln & 15);
    wsu[i] = f2bf(W[k * 64 + n]);
  } else if (i < WS_USHORTS + WS_NFLOAT) {
    const int j = i - WS_USHORTS;
    float v;
    if (j < 64)       v = b1[j];
    else if (j < 128) v = b2[j - 64];
    else if (j < 192) v = ab1[j - 128];
    else if (j < 256) v = ab2[j - 192];
    else              v = w3[j - 256];
    wsf[j] = v;
  }
}

__global__ __launch_bounds__(128) void uv_aggregator_kernel(
    const int* __restrict__ nodes, const int* __restrict__ history_uv,
    const int* __restrict__ history_r, const float* __restrict__ v2e_w,
    const float* __restrict__ u2e_w, const float* __restrict__ r2e_w,
    const unsigned short* __restrict__ wsu, const float* __restrict__ wsf,
    float* __restrict__ out) {
  __shared__ alignas(16) unsigned short sW[WS_USHORTS];  // fragment image, 48 KB
  __shared__ alignas(16) unsigned short sX[64 * 128];    // concat in / [O|uv] reuse
  __shared__ alignas(16) unsigned short sH[64 * 64];     // hidden staging
  __shared__ float sO[64 * 64];                          // o_history (f32)
  __shared__ alignas(8) unsigned int sUVp[32];           // uv packed bf16 pairs
  __shared__ float sBias[WS_NFLOAT];
  __shared__ float sLogits[64];

  const int tid = threadIdx.x;
  const int b = blockIdx.x;
  const int lane = tid & 31;
  const int wv = tid >> 5;

  // ---- copy pre-swizzled weights (b128) + bias block ----
  {
    const U4* src = reinterpret_cast<const U4*>(wsu);
    U4* dst = reinterpret_cast<U4*>(sW);
    for (int i = tid; i < WS_USHORTS / 8; i += 128) dst[i] = src[i];
    for (int i = tid; i < WS_NFLOAT; i += 128) sBias[i] = wsf[i];
  }

  if (tid < 32) {
    const float* u = u2e_w + (long)nodes[b] * 64;
    sUVp[tid] = pack_bf2(u[2 * tid], u[2 * tid + 1]);
  }

  // ---- stage X = [v2e[hist_uv] | r2e[hist_r]] as bf16, rows >= L zeroed ----
  {
    const int r = tid >> 1;
    const int seg = tid & 1;
    unsigned int* dst = reinterpret_cast<unsigned int*>(sX + r * 128 + seg * 64);
    if (r < L_HIST) {
      const float* src = (seg == 0)
          ? v2e_w + (long)history_uv[b * L_HIST + r] * 64
          : r2e_w + (long)history_r[b * L_HIST + r] * 64;
#pragma unroll
      for (int j = 0; j < 64; j += 4) {
        float4 f = *reinterpret_cast<const float4*>(src + j);
        uint2 p;
        p.x = pack_bf2(f.x, f.y);
        p.y = pack_bf2(f.z, f.w);
        *reinterpret_cast<uint2*>(dst + (j >> 1)) = p;  // ds_store_b64
      }
    } else {
      uint2 z;
      z.x = 0u;
      z.y = 0u;
#pragma unroll
      for (int j = 0; j < 32; j += 2)
        *reinterpret_cast<uint2*>(dst + j) = z;
    }
  }

  __syncthreads();

  const float* sB1 = sBias;
  const float* sB2 = sBias + 64;
  const float* sAB1 = sBias + 128;
  const float* sAB2 = sBias + 192;
  const float* sW3 = sBias + 256;

  // ---- per-wave pipeline on its 16-row M-tile ----
  unsigned short* myX = sX + wv * 16 * 128;
  unsigned short* myH = sH + wv * 16 * 64;
  float* myO = sO + wv * 16 * 64;

  // H = relu(X @ W1 + b1)
  wave_gemm<4, 0>(myX, 128, sW + WS_W1, sB1, myH, 64, nullptr, nullptr, nullptr, 0, lane);
  // O = relu(H @ W2 + b2); f32 copy for weighted sum, bf16 into sX[:, :64]
  wave_gemm<2, 1>(myH, 64, sW + WS_W2, sB2, myX, 128, myO, nullptr, nullptr, 0, lane);
  // attention concat: sX[:, 64:128] = uv_rep (b64 stores)
  {
    const uint2* uvp2 = reinterpret_cast<const uint2*>(sUVp);
    for (int i = lane; i < 16 * 16; i += 32) {
      const int m = i >> 4, c = i & 15;
      reinterpret_cast<uint2*>(myX + m * 128 + 64)[c] = uvp2[c];
    }
  }
  // A1 = relu([O|uv] @ att1 + b)
  wave_gemm<4, 0>(myX, 128, sW + WS_A1, sAB1, myH, 64, nullptr, nullptr, nullptr, 0, lane);
  // A2 = relu(A1 @ att2 + b); logits[row] = A2 . att3 (shuffle reduce, b3 cancels)
  wave_gemm<2, 2>(myH, 64, sW + WS_A2, sAB2, nullptr, 0, nullptr, sW3, sLogits, wv * 16, lane);

  __syncthreads();

  // ---- softmax over history (wave 0, shuffle reductions) ----
  if (wv == 0) {
    float v0 = (lane < L_HIST) ? sLogits[lane] : -3.0e38f;
    const int l1 = lane + 32;
    float v1 = (l1 < L_HIST) ? sLogits[l1] : -3.0e38f;
    float mx = fmaxf(v0, v1);
#pragma unroll
    for (int k = 16; k >= 1; k >>= 1) mx = fmaxf(mx, __shfl_xor(mx, k));
    float e0 = (lane < L_HIST) ? __expf(v0 - mx) : 0.f;
    float e1 = (l1 < L_HIST) ? __expf(v1 - mx) : 0.f;
    float s = e0 + e1;
#pragma unroll
    for (int k = 16; k >= 1; k >>= 1) s += __shfl_xor(s, k);
    const float inv = 1.f / s;
    sLogits[lane] = e0 * inv;
    sLogits[l1] = e1 * inv;
  }

  __syncthreads();

  // ---- out[b][d] = sum_l att[l] * O[l][d] ----
  if (tid < 64) {
    float acc = 0.f;
    for (int l = 0; l < L_HIST; ++l)
      acc = fmaf(sLogits[l], sO[l * 64 + tid], acc);
    out[(long)b * 64 + tid] = acc;
  }
}

extern "C" void kernel_launch(void* const* d_in, const int* in_sizes, int n_in,
                              void* d_out, int out_size, void* d_ws, size_t ws_size,
                              hipStream_t stream) {
  (void)in_sizes; (void)n_in; (void)out_size; (void)ws_size;
  unsigned short* wsu = reinterpret_cast<unsigned short*>(d_ws);
  float* wsf = reinterpret_cast<float*>(reinterpret_cast<char*>(d_ws) + WS_USHORTS * 2);

  const int prep_total = WS_USHORTS + WS_NFLOAT;
  prep_weights_kernel<<<(prep_total + 255) / 256, 256, 0, stream>>>(
      (const float*)d_in[6], (const float*)d_in[8],
      (const float*)d_in[10], (const float*)d_in[12],
      (const float*)d_in[7], (const float*)d_in[9],
      (const float*)d_in[11], (const float*)d_in[13],
      (const float*)d_in[14],
      wsu, wsf);

  uv_aggregator_kernel<<<B_NODES, 128, 0, stream>>>(
      (const int*)d_in[0], (const int*)d_in[1], (const int*)d_in[2],
      (const float*)d_in[3], (const float*)d_in[4], (const float*)d_in[5],
      wsu, wsf, (float*)d_out);
}
